// NodeDenoiser_50302656970807
// MI455X (gfx1250) — matrix operand order repaired and checked by reading, not compiled, not run on hardware
//
#include <hip/hip_runtime.h>
#include <hip/hip_bf16.h>
#include <math.h>

// ---------------------------------------------------------------------------
// Graph-DiT denoiser for MI455X (gfx1250): f16 WMMA for all matmuls, TDM
// (tensor_load_to_lds) for the edge-tile DMA, b128 LDS fragment traffic,
// zero-init accumulators with epilogue bias (inline C=0 WMMA).
// Z=2, N=2048, K=32 neighbors, DM=128, H=4 heads, DK=32, L=3 layers.
// ---------------------------------------------------------------------------

typedef __attribute__((ext_vector_type(16))) _Float16 v16h;
typedef __attribute__((ext_vector_type(8)))  _Float16 h8;
typedef __attribute__((ext_vector_type(8)))  float    v8f;
typedef __attribute__((ext_vector_type(4)))  unsigned int u32x4;
typedef __attribute__((ext_vector_type(8)))  int      i32x8;
typedef __attribute__((ext_vector_type(4)))  int      i32x4;

#define ZB      2
#define NN      2048
#define NNK     32
#define DMODEL  128
#define NHEAD   4
#define DHEAD   32
#define LDH     136                  // padded LDS row (halfs); 272B, 16B-aligned
#define MODSTR  (6 * DMODEL)         // per-z modulation stride (floats)

#if defined(__HIP_DEVICE_COMPILE__) && __has_builtin(__builtin_amdgcn_tensor_load_to_lds)
#define HAVE_TDM 1
#else
#define HAVE_TDM 0
#endif

__device__ __forceinline__ float silu_f(float x) { return x / (1.f + __expf(-x)); }

__device__ __forceinline__ v8f wmma16(v16h a, v16h b, v8f c) {
  // v_wmma_f32_16x16x32_f16
  return __builtin_amdgcn_wmma_f32_16x16x32_f16(false, a, false, b, (short)0, c,
                                                false, false);
}

__device__ __forceinline__ v8f zero8() {
  v8f c;
#pragma unroll
  for (int r = 0; r < 8; ++r) c[r] = 0.f;
  return c;
}

// A fragment: 16x32 f16 (rows row0.., k0..k0+31) from row-major LDS [rows][LDH].
// Lane layout (ISA): lanes 0-15 row m, K {0-7,16-23}; lanes 16-31 K {8-15,24-31}.
// Two aligned b128 LDS loads per lane.
__device__ __forceinline__ v16h lda_frag(const _Float16* A, int row0, int k0) {
  const int lane = threadIdx.x & 31;
  const int r  = row0 + (lane & 15);
  const int kb = k0 + ((lane >> 4) << 3);
  const _Float16* p = A + r * LDH + kb;
  h8 x = *(const h8*)p;
  h8 y = *(const h8*)(p + 16);
  return __builtin_shufflevector(x, y, 0, 1, 2, 3, 4, 5, 6, 7,
                                 8, 9, 10, 11, 12, 13, 14, 15);
}

// B fragment: 32x16 f16 (k0..k0+31, cols n0..n0+15) from TRANSPOSED weight LDS
// WT[n][k]. Lanes 0-15: col n, K {0-7,16-23}; lanes 16-31: col n, K {8-15,24-31}.
// Two aligned b128 LDS loads per lane.
__device__ __forceinline__ v16h ldb_frag_t(const _Float16* WT, int k0, int n0) {
  const int lane = threadIdx.x & 31;
  const int n  = n0 + (lane & 15);
  const int kb = k0 + ((lane >> 4) << 3);
  const _Float16* p = WT + n * LDH + kb;
  h8 x = *(const h8*)p;
  h8 y = *(const h8*)(p + 16);
  return __builtin_shufflevector(x, y, 0, 1, 2, 3, 4, 5, 6, 7,
                                 8, 9, 10, 11, 12, 13, 14, 15);
}

// C += A[rows][128] @ B[128][128]; B staged transposed. 32 v_wmma per call.
__device__ __forceinline__ void gemm128(const _Float16* A, const _Float16* BT,
                                        int row0, v8f acc[8]) {
#pragma unroll
  for (int kk = 0; kk < 128; kk += 32) {
    v16h a = lda_frag(A, row0, kk);
#pragma unroll
    for (int ct = 0; ct < 8; ++ct) {
      v16h b = ldb_frag_t(BT, kk, ct * 16);
      acc[ct] = wmma16(a, b, acc[ct]);
    }
  }
}

// Store C tile (16x16 f32) to LDS f16 with epilogue bias, optional silu.
__device__ __forceinline__ void stc_lds(_Float16* O, int row0, int ct, v8f c,
                                        const float* bias, bool dosilu) {
  const int lane = threadIdx.x & 31;
  const int n  = ct * 16 + (lane & 15);
  const int mb = row0 + ((lane >> 4) << 3);
  const float bv = bias[n];
#pragma unroll
  for (int r = 0; r < 8; ++r) {
    float v = c[r] + bv;
    if (dosilu) v = silu_f(v);
    O[(mb + r) * LDH + n] = (_Float16)v;
  }
}

// Stage a 128x128 f32 weight sub-block TRANSPOSED into LDS f16: WT[n][k].
// One b128 LDS store per 8 k's; global reads coalesced over n.
__device__ __forceinline__ void load_w_t(const float* W, int ncols_total,
                                         int nstart, _Float16* WT, int nthr) {
  for (int i = threadIdx.x; i < 128 * 16; i += nthr) {
    const int n  = i & 127;
    const int kb = (i >> 7) << 3;
    h8 v;
#pragma unroll
    for (int j = 0; j < 8; ++j)
      v[j] = (_Float16)W[(kb + j) * ncols_total + nstart + n];
    *(h8*)(WT + n * LDH + kb) = v;
  }
}

// Head-concatenated projection weight [H][DM][DK] -> transposed WT[n=h*DK+e][k].
__device__ __forceinline__ void load_wcat_t(const float* W, _Float16* WT,
                                            int nthr) {
  for (int i = threadIdx.x; i < 128 * 16; i += nthr) {
    const int n  = i & 127;
    const int kb = (i >> 7) << 3;
    const int h = n >> 5, e = n & 31;
    h8 v;
#pragma unroll
    for (int j = 0; j < 8; ++j)
      v[j] = (_Float16)W[(h * DMODEL + (kb + j)) * DHEAD + e];
    *(h8*)(WT + n * LDH + kb) = v;
  }
}

// ---------------------------------------------------------------------------
// adaLN kernel: tiny MLPs on t (one block per z). Reproduces the swapped
// unpack: alpha = gb[0:128], gamma = gb[128:256], beta = a-MLP output.
// mod layout per z: [0]=gamma1 [1]=beta1 [2]=alpha1 [3]=gamma2 [4]=beta2 [5]=alpha2
// ---------------------------------------------------------------------------
__device__ void mlp3_eval(const float* x, const float* W0, const float* b0,
                          const float* W1, const float* b1, const float* W2,
                          const float* b2, int lastN, float* h1, float* h2,
                          float* out) {
  const int tid = threadIdx.x;
  if (tid < 128) {
    float a = b0[tid];
    for (int d = 0; d < 128; ++d) a += x[d] * W0[d * 128 + tid];
    h1[tid] = silu_f(a);
  }
  __syncthreads();
  if (tid < 128) {
    float a = b1[tid];
    for (int d = 0; d < 128; ++d) a += h1[d] * W1[d * 128 + tid];
    h2[tid] = silu_f(a);
  }
  __syncthreads();
  if (tid < lastN) {
    float a = b2[tid];
    for (int d = 0; d < 128; ++d) a += h2[d] * W2[d * lastN + tid];
    out[tid] = a;
  }
  __syncthreads();
}

__launch_bounds__(256)
__global__ void adaln_kernel(const float* t,
    const float* agW0, const float* agb0, const float* agW1, const float* agb1,
    const float* agW2, const float* agb2,
    const float* aaW0, const float* aab0, const float* aaW1, const float* aab1,
    const float* aaW2, const float* aab2,
    const float* fgW0, const float* fgb0, const float* fgW1, const float* fgb1,
    const float* fgW2, const float* fgb2,
    const float* faW0, const float* fab0, const float* faW1, const float* fab1,
    const float* faW2, const float* fab2,
    float* modb) {
  __shared__ float tin[128], h1[128], h2[128], outb[256];
  const int z = blockIdx.x, tid = threadIdx.x;
  if (tid < 128) tin[tid] = t[z * DMODEL + tid];
  __syncthreads();
  float* m = modb + z * MODSTR;
  mlp3_eval(tin, agW0, agb0, agW1, agb1, agW2, agb2, 256, h1, h2, outb);
  if (tid < 128) { m[2 * 128 + tid] = outb[tid]; m[0 * 128 + tid] = outb[128 + tid]; }
  __syncthreads();
  mlp3_eval(tin, aaW0, aab0, aaW1, aab1, aaW2, aab2, 128, h1, h2, outb);
  if (tid < 128) m[1 * 128 + tid] = outb[tid];
  __syncthreads();
  mlp3_eval(tin, fgW0, fgb0, fgW1, fgb1, fgW2, fgb2, 256, h1, h2, outb);
  if (tid < 128) { m[5 * 128 + tid] = outb[tid]; m[3 * 128 + tid] = outb[128 + tid]; }
  __syncthreads();
  mlp3_eval(tin, faW0, fab0, faW1, fab1, faW2, fab2, 128, h1, h2, outb);
  if (tid < 128) m[4 * 128 + tid] = outb[tid];
}

// ---------------------------------------------------------------------------
// prep: nodes_i = gamma1 * static_ln(nodes) + beta1   (f16 out, one thread/row)
// ---------------------------------------------------------------------------
__launch_bounds__(256)
__global__ void prep_kernel(const float* nodes, const float* modb, _Float16* NIb) {
  const int row = blockIdx.x * 256 + threadIdx.x;          // 0..4095
  if (row >= ZB * NN) return;
  const int z = row >> 11;
  const float* xr = nodes + (size_t)row * DMODEL;
  float s = 0.f, ss = 0.f;
  for (int c = 0; c < 128; ++c) { float v = xr[c]; s += v; ss += v * v; }
  const float mean = s * (1.f / 128.f);
  float var = (ss - 128.f * mean * mean) * (1.f / 127.f);   // ddof=1
  float sd = sqrtf(fmaxf(var, 0.f));
  if (sd == 0.f) sd = 1.f;
  const float inv = 1.f / sd;
  const float* g = modb + z * MODSTR;            // gamma1
  const float* b = modb + z * MODSTR + 128;      // beta1
  for (int c = 0; c < 128; ++c)
    NIb[(size_t)row * DMODEL + c] = (_Float16)(g[c] * ((xr[c] - mean) * inv) + b[c]);
}

// ---------------------------------------------------------------------------
// qproj: Q = nodes_i @ qWcat + qb   (WMMA GEMM, 128 rows per block, 8 waves)
// ---------------------------------------------------------------------------
__launch_bounds__(256)
__global__ void qproj_kernel(const _Float16* NIb, const float* qW, const float* qb,
                             float* Qb) {
  __shared__ __align__(16) _Float16 XB[128 * LDH];
  __shared__ __align__(16) _Float16 WB[128 * LDH];
  const int tid = threadIdx.x;
  const int rows0 = blockIdx.x * 128;
  // b128 global->LDS copy of 128 rows of nodes_i
  for (int i = tid; i < 128 * 16; i += 256) {
    const int r = i >> 4, c8 = (i & 15) << 3;
    *(h8*)(XB + r * LDH + c8) =
        *(const h8*)(NIb + (size_t)(rows0 + r) * DMODEL + c8);
  }
  load_wcat_t(qW, WB, 256);
  __syncthreads();
  const int row0 = (tid >> 5) * 16;
  v8f acc[8];
#pragma unroll
  for (int ct = 0; ct < 8; ++ct) acc[ct] = zero8();
  gemm128(XB, WB, row0, acc);
  const int lane = tid & 31;
#pragma unroll
  for (int ct = 0; ct < 8; ++ct) {
    const int n  = ct * 16 + (lane & 15);
    const int mb = row0 + ((lane >> 4) << 3);
    const float bv = qb[n];
#pragma unroll
    for (int r = 0; r < 8; ++r)
      Qb[(size_t)(rows0 + mb + r) * DMODEL + n] = acc[ct][r] + bv;
  }
}

// ---------------------------------------------------------------------------
// edge+attention kernel: one block per 2 nodes (64 edge rows), 4 waves.
// FiLM(edges) 3-layer MLP + K/V head projection on WMMA, scalar attention.
// ---------------------------------------------------------------------------
__device__ void film_proj(const float* W1, const float* b1, const float* W2,
                          const float* b2, const float* W3, const float* b3,
                          const float* pW, const float* pb,
                          _Float16* XB, _Float16* NJ, _Float16* WB,
                          _Float16* H1, _Float16* H2, _Float16* OUTP) {
  const int tid = threadIdx.x;
  const int row0 = (tid >> 5) * 16;     // 4 waves x 16 rows = 64 edge rows
  // H1 = silu(X @ W1 + b1)
  load_w_t(W1, 128, 0, WB, 128);
  __syncthreads();
  {
    v8f acc[8];
#pragma unroll
    for (int ct = 0; ct < 8; ++ct) acc[ct] = zero8();
    gemm128(XB, WB, row0, acc);
#pragma unroll
    for (int ct = 0; ct < 8; ++ct) stc_lds(H1, row0, ct, acc[ct], b1, true);
  }
  __syncthreads();
  // H2 = silu(H1 @ W2 + b2)
  load_w_t(W2, 128, 0, WB, 128);
  __syncthreads();
  {
    v8f acc[8];
#pragma unroll
    for (int ct = 0; ct < 8; ++ct) acc[ct] = zero8();
    gemm128(H1, WB, row0, acc);
#pragma unroll
    for (int ct = 0; ct < 8; ++ct) stc_lds(H2, row0, ct, acc[ct], b2, true);
  }
  __syncthreads();
  // gamma half of FiLM head (cols 0..127 of W3 [128][256]), kept in registers
  load_w_t(W3, 256, 0, WB, 128);
  __syncthreads();
  v8f g[8];
#pragma unroll
  for (int ct = 0; ct < 8; ++ct) g[ct] = zero8();
  gemm128(H2, WB, row0, g);
  __syncthreads();
  // beta half (cols 128..255), fuse: KF = (g+bg) * nodes_j + (bb+bbv) -> H1
  load_w_t(W3, 256, 128, WB, 128);
  __syncthreads();
  {
    v8f bb[8];
#pragma unroll
    for (int ct = 0; ct < 8; ++ct) bb[ct] = zero8();
    gemm128(H2, WB, row0, bb);
    const int lane = tid & 31;
#pragma unroll
    for (int ct = 0; ct < 8; ++ct) {
      const int n  = ct * 16 + (lane & 15);
      const int mb = row0 + ((lane >> 4) << 3);
      const float bg  = b3[n];
      const float bbv = b3[128 + n];
#pragma unroll
      for (int r = 0; r < 8; ++r) {
        float kf = (g[ct][r] + bg) * (float)NJ[(mb + r) * LDH + n]
                   + (bb[ct][r] + bbv);
        H1[(mb + r) * LDH + n] = (_Float16)kf;
      }
    }
  }
  __syncthreads();
  // head projection: OUTP = KF @ pWcat + pb
  load_wcat_t(pW, WB, 128);
  __syncthreads();
  {
    v8f acc[8];
#pragma unroll
    for (int ct = 0; ct < 8; ++ct) acc[ct] = zero8();
    gemm128(H1, WB, row0, acc);
#pragma unroll
    for (int ct = 0; ct < 8; ++ct) stc_lds(OUTP, row0, ct, acc[ct], pb, false);
  }
  __syncthreads();
}

__launch_bounds__(128)
__global__ void edge_attn_kernel(float* nodes, const float* edges, const int* nbrs,
    const unsigned char* nmask, const _Float16* NIb, const float* Qb,
    const float* modb,
    const float* fKW0, const float* fKb0, const float* fKW1, const float* fKb1,
    const float* fKW2, const float* fKb2,
    const float* fVW0, const float* fVb0, const float* fVW1, const float* fVb1,
    const float* fVW2, const float* fVb2,
    const float* kW, const float* kb, const float* vW, const float* vb,
    const float* oW) {
  __shared__ __align__(16) _Float16 XB[64 * LDH];   // edges (f16)
  __shared__ __align__(16) _Float16 NJ[64 * LDH];   // gathered neighbor nodes_i
  __shared__ __align__(16) _Float16 WB[128 * LDH];  // weights / TDM f32 staging
  __shared__ __align__(16) _Float16 H1[64 * LDH];
  __shared__ __align__(16) _Float16 H2[64 * LDH];
  __shared__ __align__(16) _Float16 KP[64 * LDH];   // K head features [64][H*DK]
  __shared__ __align__(16) _Float16 VP[64 * LDH];   // V head features
  __shared__ float QT[2 * DMODEL];
  __shared__ float SP[2 * DMODEL];    // scores/probs [nl][h][k]
  __shared__ float AO[2 * DMODEL];    // attn out pre-oW, d = e*H + h
  __shared__ int   NBs[64];
  __shared__ unsigned char MBs[64];

  const int tid = threadIdx.x;
  const int wg  = blockIdx.x;
  const int z   = wg / (NN / 2);
  const int n0  = (wg % (NN / 2)) * 2;

  if (tid < 64) {
    const int gidx = (z * NN + n0 + (tid >> 5)) * NNK + (tid & 31);
    NBs[tid] = nbrs[gidx];
    MBs[tid] = nmask[gidx];
  }
  const float* eb = edges + (size_t)(z * NN + n0) * NNK * DMODEL;
  float* WBf = (float*)WB;            // 64*128 f32 = 32KB fits in WB (34.8KB)

#if HAVE_TDM
  // Tensor Data Mover: DMA the 64x128 f32 edge tile into LDS (WB staging).
  // D# per CDNA5 ISA ch.8: group0 = {count|flags, lds_addr, gaddr_lo, gaddr_hi|type}
  if (tid < 32) {                      // one wave issues the TDM op
    const unsigned long long ga = (unsigned long long)(uintptr_t)eb;
    u32x4 g0;
    g0[0] = 1u;                                        // count=1, user D#
    g0[1] = (unsigned)(uintptr_t)WBf;                  // lds_addr (low 32 bits)
    g0[2] = (unsigned)(ga & 0xFFFFFFFFu);              // global_addr[31:0]
    g0[3] = (unsigned)((ga >> 32) & 0x01FFFFFFu) | (2u << 30);  // addr[56:32],type=2
    i32x8 g1;
    g1[0] = 0x20000;                                   // data_size=2 (4B)
    g1[1] = (int)(128u << 16);                         // tensor_dim0 = 128
    g1[2] = (int)(64u << 16);                          // tensor_dim1 = 64
    g1[3] = (int)(128u << 16);                         // tile_dim0 = 128
    g1[4] = 64;                                        // tile_dim1 = 64
    g1[5] = 128;                                       // tensor_dim0_stride = 128
    g1[6] = 0;
    g1[7] = 0;
    i32x4 z4; z4[0] = 0; z4[1] = 0; z4[2] = 0; z4[3] = 0;
#if defined(__clang_major__) && __clang_major__ >= 23
    i32x8 z8; for (int j = 0; j < 8; ++j) z8[j] = 0;
    __builtin_amdgcn_tensor_load_to_lds(g0, g1, z4, z4, z8, 0);
#else
    __builtin_amdgcn_tensor_load_to_lds(g0, g1, z4, z4, 0);
#endif
  }
#else
  for (int i = tid; i < 64 * 128; i += 128) WBf[i] = eb[i];
#endif
  // pull fV weights toward L2 while fK runs
  if ((tid & 31) == 0) {
    __builtin_prefetch(fVW0, 0, 1);
    __builtin_prefetch(fVW1, 0, 1);
    __builtin_prefetch(fVW2, 0, 1);
    __builtin_prefetch(vW, 0, 1);
  }
  __syncthreads();
  // gather neighbor features while the TDM streams (b128 global + b128 LDS)
  for (int i = tid; i < 64 * 16; i += 128) {
    const int r = i >> 4, c8 = (i & 15) << 3;
    *(h8*)(NJ + r * LDH + c8) =
        *(const h8*)(NIb + (size_t)(z * NN + NBs[r]) * DMODEL + c8);
  }
#if HAVE_TDM
  if (tid < 32) __builtin_amdgcn_s_wait_tensorcnt(0);
#endif
  __syncthreads();
  // convert staged f32 edges -> f16 WMMA layout
  for (int i = tid; i < 64 * 128; i += 128)
    XB[(i >> 7) * LDH + (i & 127)] = (_Float16)WBf[i];
  __syncthreads();

  film_proj(fKW0, fKb0, fKW1, fKb1, fKW2, fKb2, kW, kb, XB, NJ, WB, H1, H2, KP);
  film_proj(fVW0, fVb0, fVW1, fVb1, fVW2, fVb2, vW, vb, XB, NJ, WB, H1, H2, VP);

  // ---- attention for the 2 center nodes -----------------------------------
  for (int i = tid; i < 2 * DMODEL; i += 128)
    QT[i] = Qb[(size_t)(z * NN + n0 + (i >> 7)) * DMODEL + (i & 127)];
  __syncthreads();
  // scores S[nl][h][k]
  for (int i = tid; i < 2 * DMODEL; i += 128) {
    const int nl = i >> 7, rem = i & 127, h = rem >> 5, k = rem & 31;
    const _Float16* kp = &KP[(nl * 32 + k) * LDH + h * 32];
    const float* q = &QT[nl * 128 + h * 32];
    float s = 0.f;
#pragma unroll
    for (int e = 0; e < 32; ++e) s += q[e] * (float)kp[e];
    SP[i] = s * 0.17677669529663687f;   // 1/sqrt(32)
  }
  __syncthreads();
  // masked softmax over k (don't mask fully-invalid rows)
  if (tid < 8) {
    const int nl = tid >> 2, h = tid & 3;
    const unsigned char* mrow = &MBs[nl * 32];
    bool any = false;
    for (int k = 0; k < 32; ++k) any = any || (mrow[k] != 0);
    float* sp = &SP[nl * 128 + h * 32];
    float mx = -1e30f;
    for (int k = 0; k < 32; ++k)
      if (mrow[k] || !any) mx = fmaxf(mx, sp[k]);
    float sum = 0.f;
    for (int k = 0; k < 32; ++k) {
      const float p = (mrow[k] || !any) ? __expf(sp[k] - mx) : 0.f;
      sp[k] = p; sum += p;
    }
    const float inv = 1.f / sum;
    for (int k = 0; k < 32; ++k) sp[k] *= inv;
  }
  __syncthreads();
  // out[nl][d], d = e*H + h  (torch permute(0,2,3,1).reshape)
  for (int i = tid; i < 2 * DMODEL; i += 128) {
    const int nl = i >> 7, d = i & 127, e = d >> 2, h = d & 3;
    const float* p = &SP[nl * 128 + h * 32];
    float o = 0.f;
#pragma unroll
    for (int k = 0; k < 32; ++k) o += p[k] * (float)VP[(nl * 32 + k) * LDH + h * 32 + e];
    AO[i] = o;
  }
  __syncthreads();
  // o-projection + alpha1-scaled residual
  const float* a1 = modb + z * MODSTR + 2 * 128;
  for (int i = tid; i < 2 * DMODEL; i += 128) {
    const int nl = i >> 7, c = i & 127;
    const float* ar = &AO[nl * 128];
    float acc = 0.f;
    for (int d = 0; d < 128; ++d) acc += ar[d] * oW[d * 128 + c];
    const size_t gi = (size_t)(z * NN + n0 + nl) * DMODEL + c;
    nodes[gi] = nodes[gi] + a1[c] * acc;
  }
}

// ---------------------------------------------------------------------------
// ffn: nodes += alpha2 * MLP(gamma2*ln(nodes)+beta2); hidden 512 in 4 chunks.
// ---------------------------------------------------------------------------
__launch_bounds__(256)
__global__ void ffn_kernel(float* nodes, const float* modb, const float* W1,
                           const float* b1, const float* W2, const float* b2) {
  __shared__ __align__(16) _Float16 XB[128 * LDH];
  __shared__ __align__(16) _Float16 WB[128 * LDH];
  __shared__ __align__(16) _Float16 HB[128 * LDH];
  const int tid = threadIdx.x;
  const int rows0 = blockIdx.x * 128;
  const int z = rows0 >> 11;
  const float* g2  = modb + z * MODSTR + 3 * 128;
  const float* be2 = modb + z * MODSTR + 4 * 128;
  const float* a2  = modb + z * MODSTR + 5 * 128;
  if (tid < 128) {
    const int row = rows0 + tid;
    const float* xr = nodes + (size_t)row * DMODEL;
    float s = 0.f, ss = 0.f;
    for (int c = 0; c < 128; ++c) { float v = xr[c]; s += v; ss += v * v; }
    const float mean = s * (1.f / 128.f);
    float var = (ss - 128.f * mean * mean) * (1.f / 127.f);
    float sd = sqrtf(fmaxf(var, 0.f));
    if (sd == 0.f) sd = 1.f;
    const float inv = 1.f / sd;
    for (int c = 0; c < 128; ++c)
      XB[tid * LDH + c] = (_Float16)(g2[c] * ((xr[c] - mean) * inv) + be2[c]);
  }
  const int row0 = (tid >> 5) * 16;
  v8f out[8];
#pragma unroll
  for (int ct = 0; ct < 8; ++ct) out[ct] = zero8();
  for (int c = 0; c < 4; ++c) {
    __syncthreads();
    load_w_t(W1, 512, c * 128, WB, 256);
    __syncthreads();
    v8f h[8];
#pragma unroll
    for (int ct = 0; ct < 8; ++ct) h[ct] = zero8();
    gemm128(XB, WB, row0, h);
#pragma unroll
    for (int ct = 0; ct < 8; ++ct) stc_lds(HB, row0, ct, h[ct], b1 + c * 128, true);
    __syncthreads();
    load_w_t(W2 + (size_t)c * 128 * 128, 128, 0, WB, 256);
    __syncthreads();
    gemm128(HB, WB, row0, out);
  }
  const int lane = tid & 31;
#pragma unroll
  for (int ct = 0; ct < 8; ++ct) {
    const int n  = ct * 16 + (lane & 15);
    const int mb = row0 + ((lane >> 4) << 3);
    const float av = a2[n];
    const float bv = b2[n];
#pragma unroll
    for (int r = 0; r < 8; ++r) {
      const size_t gi = (size_t)(rows0 + mb + r) * DMODEL + n;
      nodes[gi] += av * (out[ct][r] + bv);
    }
  }
}

// ---------------------------------------------------------------------------
// launch
// ---------------------------------------------------------------------------
extern "C" void kernel_launch(void* const* d_in, const int* in_sizes, int n_in,
                              void* d_out, int out_size, void* d_ws, size_t ws_size,
                              hipStream_t stream) {
  (void)in_sizes; (void)n_in; (void)out_size; (void)ws_size;
  const float* nodes_in = (const float*)d_in[0];
  const float* t        = (const float*)d_in[1];
  const float* edges    = (const float*)d_in[2];
  const int*   nbrs     = (const int*)d_in[3];
  const unsigned char* nmask = (const unsigned char*)d_in[4];

  float*    nodes = (float*)d_out;                 // working copy lives in d_out
  float*    modb  = (float*)d_ws;                  // [Z][6][128]
  float*    Qb    = modb + ZB * MODSTR;            // [Z*N][128] f32
  _Float16* NIb   = (_Float16*)(Qb + (size_t)ZB * NN * DMODEL);  // [Z*N][128] f16

  hipMemcpyAsync(nodes, nodes_in, (size_t)ZB * NN * DMODEL * sizeof(float),
                 hipMemcpyDeviceToDevice, stream);

  for (int l = 0; l < 3; ++l) {
    // params flattened jax-pytree style (sorted keys) per layer, 47 leaves:
    // 0-2 an_a_Ws, 3-5 an_a_bs, 6-8 an_gb_Ws, 9-11 an_gb_bs, 12-14 fK_Ws,
    // 15-17 fK_bs, 18-20 fV_Ws, 21-23 fV_bs, 24-25 ffn_Ws, 26-27 ffn_bs,
    // 28-30 fn_a_Ws, 31-33 fn_a_bs, 34-36 fn_gb_Ws, 37-39 fn_gb_bs,
    // 40 kW, 41 kb, 42 oW, 43 qW, 44 qb, 45 vW, 46 vb
    const int base = 5 + 47 * l;
    auto P = [&](int i) { return (const float*)d_in[base + i]; };

    adaln_kernel<<<ZB, 256, 0, stream>>>(t,
        P(6), P(9), P(7), P(10), P(8), P(11),        // an_gb
        P(0), P(3), P(1), P(4), P(2), P(5),          // an_a
        P(34), P(37), P(35), P(38), P(36), P(39),    // fn_gb
        P(28), P(31), P(29), P(32), P(30), P(33),    // fn_a
        modb);

    prep_kernel<<<(ZB * NN) / 256, 256, 0, stream>>>(nodes, modb, NIb);

    qproj_kernel<<<(ZB * NN) / 128, 256, 0, stream>>>(NIb, P(43), P(44), Qb);

    edge_attn_kernel<<<ZB * (NN / 2), 128, 0, stream>>>(nodes, edges, nbrs, nmask,
        NIb, Qb, modb,
        P(12), P(15), P(13), P(16), P(14), P(17),    // fK MLP
        P(18), P(21), P(19), P(22), P(20), P(23),    // fV MLP
        P(40), P(41), P(45), P(46), P(42));          // kW, kb, vW, vb, oW

    ffn_kernel<<<(ZB * NN) / 128, 256, 0, stream>>>(nodes, modb, P(24), P(26),
                                                    P(25), P(27));
  }
}
